// BiLSTM2D_21689584845165
// MI455X (gfx1250) — matrix operand order repaired and checked by reading, not compiled, not run on hardware
//
#include <hip/hip_runtime.h>

// ---------------------------------------------------------------------------
// BiLSTM2D for MI455X (gfx1250).
// - Input-gate conv hoisted out of the recurrence: one batched bf16 WMMA GEMM
//   (164 of ~185 GFLOP) using v_wmma_f32_16x16x32_bf16 (f32 accumulate).
// - Activation tiles staged into LDS by the Tensor Data Mover with row
//   padding (pad_enable), so the im2col gather is branch-free.
// - Each wave feeds 4 WMMAs per B-gather (4 M-tiles), 2 column groups/block.
// - Recurrent W_hh GEMM + LSTM pointwise: 32 small sequential launches.
// ---------------------------------------------------------------------------

typedef __attribute__((ext_vector_type(16))) __bf16 v16bf;
typedef __attribute__((ext_vector_type(8)))  float  v8f;

union Frag {
  v16bf v;
  unsigned short s[16];
  uint4 q[2];
};
union OTab16 {
  uint4 q[4];
  unsigned u[16];
};

// Problem constants
static constexpr int Bb   = 4;
static constexpr int C    = 64;
static constexpr int T    = 256;
static constexpr int Ff   = 128;
static constexpr int WINW = 8;
static constexpr int HID  = 64;
static constexpr int NW   = 32;          // n_win = T/WIN
static constexpr int Nn   = 32;          // B*WIN
static constexpr int L    = 61;          // (F-8)/2+1
static constexpr int COLS = Nn * L;      // 1952
static constexpr int G4   = 256;         // 4*HID
static constexpr int CW   = 512;         // C*WIN (per direction)
static constexpr int KIH  = CW * 5;      // 2560
static constexpr int KT_IH = KIH / 32;   // 80 k-tiles
static constexpr int MT   = G4 / 16;     // 16 m-tiles
static constexpr int NT32 = COLS / 32;   // 61 column tiles of 32

// Padded LDS activation slice: 16-ushort front pad + 64 rows of (128 data +
// 16 pad) ushorts. Any gather address 16 + row*144 + j + 2*lp, lp in [-2,62],
// stays inside [12, 9232).
static constexpr int ROWP  = 144;
static constexpr int SLICE = 16 + 64 * ROWP;   // 9232 ushorts

__device__ __forceinline__ unsigned short f2bf(float f) {
  unsigned u = __float_as_uint(f);
  unsigned r = u + 0x7fffu + ((u >> 16) & 1u);   // round-to-nearest-even
  return (unsigned short)(r >> 16);
}
__device__ __forceinline__ float bf2f(unsigned short s) {
  return __uint_as_float(((unsigned)s) << 16);
}
__device__ __forceinline__ float sigm(float x) { return 1.0f / (1.0f + __expf(-x)); }
__device__ __forceinline__ float tanh_f(float x) {
  float cx = fminf(fmaxf(x, -15.0f), 15.0f);
  float e = __expf(2.0f * cx);
  return (e - 1.0f) / (e + 1.0f);
}

#if __has_builtin(__builtin_amdgcn_tensor_load_to_lds)
#define HAS_TDM 1
#else
#define HAS_TDM 0
#endif

#if HAS_TDM
// One TDM descriptor: 1D tile of 8192 bf16 elements, LDS padding of 8 DWORDs
// (32B) after every 64 DWORDs (256B) -> 288B LDS row stride.
__device__ __forceinline__ void tdm_load_slice(unsigned lds_byte_addr, const void* gsrc) {
  typedef unsigned v4u __attribute__((ext_vector_type(4)));
  typedef int      v4i __attribute__((ext_vector_type(4)));
  typedef int      v8i __attribute__((ext_vector_type(8)));
  unsigned long long ga = (unsigned long long)(size_t)gsrc;
  v4u g0 = { 1u,                                   // count=1 (valid descriptor)
             lds_byte_addr,                        // lds_addr
             (unsigned)ga,                         // global_addr[31:0]
             ((unsigned)(ga >> 32) & 0x01ffffffu) | 0x80000000u }; // addr[56:32] | type=2
  v8i g1 = { (int)((1u << 16)      // data_size = 2 bytes
                 | (1u << 20)      // pad_enable
                 | (5u << 22)      // pad_interval: 2^6 = 64 DWORDs (256B)
                 | (7u << 25)),    // pad_amount: 8 DWORDs (32B)
             (int)0x20000000,      // tensor_dim0 = 8192 (bits[63:48])
             0,                    // tensor_dim0 hi / tensor_dim1 lo
             (int)0x20000000,      // tile_dim0 = 8192 (bits[127:112])
             0, 0, 0, 0 };
  v4i gz = { 0, 0, 0, 0 };
#if defined(__clang_major__) && __clang_major__ >= 23
  v8i gz8 = { 0, 0, 0, 0, 0, 0, 0, 0 };
  __builtin_amdgcn_tensor_load_to_lds(g0, g1, gz, gz, gz8, 0);
#else
  __builtin_amdgcn_tensor_load_to_lds(g0, g1, gz, gz, 0);
#endif
}
#endif

// ---------------------------------------------------------------------------
// 1) gLN statistics: per-batch sum / sumsq over C*T*F = 2,097,152 elements
// ---------------------------------------------------------------------------
__global__ void k_stats(const float* __restrict__ x, float* __restrict__ stats) {
  __shared__ float ss[256], sq[256];
  int b = blockIdx.y;
  size_t base = (size_t)b * 2097152 + (size_t)blockIdx.x * 2048;
  float s = 0.f, q = 0.f;
#pragma unroll
  for (int r = 0; r < 8; ++r) {
    float v = x[base + r * 256 + threadIdx.x];
    s += v; q += v * v;
  }
  ss[threadIdx.x] = s; sq[threadIdx.x] = q;
  __syncthreads();
  for (int st = 128; st > 0; st >>= 1) {
    if ((int)threadIdx.x < st) {
      ss[threadIdx.x] += ss[threadIdx.x + st];
      sq[threadIdx.x] += sq[threadIdx.x + st];
    }
    __syncthreads();
  }
  if (threadIdx.x == 0) {
    atomicAdd(&stats[b * 2 + 0], ss[0]);
    atomicAdd(&stats[b * 2 + 1], sq[0]);
  }
}

// ---------------------------------------------------------------------------
// 2) Normalize + build scan input xs[t][n][c2][f] (bf16)
// ---------------------------------------------------------------------------
__global__ void k_xs(const float* __restrict__ x, const float* __restrict__ g,
                     const float* __restrict__ be, const float* __restrict__ stats,
                     unsigned short* __restrict__ xs) {
  int blk = blockIdx.x;            // t*32 + n
  int t = blk >> 5, n = blk & 31;
  int b = n >> 3, w = n & 7;
  const float inv = 1.0f / 2097152.0f;
  float m = stats[b * 2] * inv;
  float var = stats[b * 2 + 1] * inv - m * m;
  float rstd = rsqrtf(var + 1e-8f);
#pragma unroll 4
  for (int r = 0; r < 64; ++r) {
    int e = r * 256 + threadIdx.x;
    int c2 = e >> 7, f = e & 127;
    int c = c2 & 63;
    int tg = t * 8 + w;
    if (c2 >= 64) tg = (T - 1) - tg;
    float v = x[(((size_t)b * C + c) * T + tg) * Ff + f];
    float val = (v - m) * rstd * g[c] + be[c];
    xs[(size_t)blk * 16384 + e] = f2bf(val);
  }
}

// ---------------------------------------------------------------------------
// 3) Pre-swizzle weights into WMMA A fragments (contiguous k = 16*(lane/16) + e).
// ---------------------------------------------------------------------------
__global__ void k_swz_ih(const float* __restrict__ Wf, const float* __restrict__ Wb,
                         unsigned short* __restrict__ A) {
  int idx = blockIdx.x * 256 + threadIdx.x;           // < 2*16*80*32*16
  int e = idx & 15, lane = (idx >> 4) & 31;
  int rest = idx >> 9;
  int kt = rest % KT_IH;
  int rest2 = rest / KT_IH;
  int mt = rest2 & 15, d = rest2 >> 4;
  int koff = ((lane < 16) ? 0 : 16) + e;
  int Kg = kt * 32 + koff;
  int M = mt * 16 + (lane & 15);
  int i5 = Kg / 5, tap = Kg - i5 * 5;                 // K = channel*5 + tap
  const float* W = d ? Wb : Wf;                       // (256, 512, 5)
  A[idx] = f2bf(W[((size_t)M * CW + i5) * 5 + tap]);
}

__global__ void k_swz_hh(const float* __restrict__ Wf, const float* __restrict__ Wb,
                         unsigned short* __restrict__ A) {
  int idx = blockIdx.x * 256 + threadIdx.x;           // < 2*16*2*32*16
  int e = idx & 15, lane = (idx >> 4) & 31;
  int rest = idx >> 9;
  int kt = rest & 1;
  int rest2 = rest >> 1;
  int mt = rest2 & 15, d = rest2 >> 4;
  int koff = ((lane < 16) ? 0 : 16) + e;
  int Kg = kt * 32 + koff;                            // hidden index < 64
  int M = mt * 16 + (lane & 15);
  const float* W = d ? Wb : Wf;                       // (256, 64, 1)
  A[idx] = f2bf(W[(size_t)M * HID + Kg]);
}

// ---------------------------------------------------------------------------
// 4) Batched input-gate GEMM (all 32 steps x 2 dirs at once).
//    gI[t][d][256][1952] = W_ih (256 x 2560) * im2col(xs) (2560 x 1952)
//    2 column groups x 4 waves; each wave: 4 M-tiles -> 4 WMMAs per B-gather.
// ---------------------------------------------------------------------------
__global__ void __launch_bounds__(256) k_gates(const unsigned short* __restrict__ xs,
                                               const unsigned short* __restrict__ Aih,
                                               float* __restrict__ gI) {
  __shared__ alignas(16) unsigned short sX[2][SLICE]; // two padded n-slices
  __shared__ alignas(16) unsigned otab[KIH];          // gather offsets + tap
  const int nt = blockIdx.x, t = blockIdx.y, d = blockIdx.z;
  const int tid = threadIdx.x;
  const int col0 = nt * 32;
  const int n0 = col0 / L;
  const int n1 = (col0 + 31) / L;                     // 32 cols span <= 2 n's

  // Offset table: K = i5*5 + tap, i5 = c*8 + j ->
  //   off = c*ROWP + j + 2*tap (14 bits), tap in bits [18:16].
  for (int i = tid; i < KIH; i += 256) {
    int i5 = i / 5, tap = i - i5 * 5;
    otab[i] = (unsigned)((i5 >> 3) * ROWP + (i5 & 7) + 2 * tap) | ((unsigned)tap << 16);
  }

  const unsigned short* src0 = xs + (((size_t)t * Nn + n0) * 128 + d * 64) * 128;
  const unsigned short* src1 = xs + (((size_t)t * Nn + n1) * 128 + d * 64) * 128;
#if HAS_TDM
  if (tid < 32) {                                     // one wave drives the TDM
    tdm_load_slice((unsigned)(size_t)(void*)&sX[0][16], src0);
    tdm_load_slice((unsigned)(size_t)(void*)&sX[1][16], src1);
#if __has_builtin(__builtin_amdgcn_s_wait_tensorcnt)
    __builtin_amdgcn_s_wait_tensorcnt(0);
#endif
  }
#else
  {
    const uint4* s0 = (const uint4*)src0;
    const uint4* s1 = (const uint4*)src1;
#pragma unroll
    for (int r = 0; r < 4; ++r) {
      int i = tid + r * 256;                          // < 1024 uint4
      int row = i >> 4, off = i & 15;                 // 16 uint4 per 128-ushort row
      ((uint4*)&sX[0][16 + row * ROWP])[off] = s0[i];
      ((uint4*)&sX[1][16 + row * ROWP])[off] = s1[i];
    }
  }
#endif
  __syncthreads();

  const int wave = tid >> 5, lane = tid & 31;
  const int grp = wave >> 2;                          // column group (0/1)
  const int mtb = (wave & 3) * 4;                     // 4 M-tiles per wave
  const int col = col0 + grp * 16 + (lane & 15);
  const int n = col / L, l = col % L;
  const unsigned short* sxp = &sX[(n == n0) ? 0 : 1][0];
  const int kbase = (lane < 16) ? 0 : 16;             // B layout: lanes 0-15 K0-15
  const int lbase = 2 * l + 12;                       // front pad 16, conv pad -2

  v8f acc0 = {}, acc1 = {}, acc2 = {}, acc3 = {};
  for (int kt = 0; kt < KT_IH; ++kt) {
    OTab16 ot;
    {
      const uint4* op = (const uint4*)&otab[kt * 32 + kbase];
      ot.q[0] = op[0]; ot.q[1] = op[1]; ot.q[2] = op[2]; ot.q[3] = op[3];
    }
    Frag bf_;
#pragma unroll
    for (int e = 0; e < 16; ++e) {
      unsigned ov = ot.u[e];
      int tap = (int)(ov >> 16);
      int lp = l + tap - 2;
      unsigned short v = sxp[(ov & 0xffffu) + lbase]; // always in-bounds
      bf_.s[e] = ((unsigned)lp < 61u) ? v : (unsigned short)0;
    }
    Frag a0, a1, a2, a3;
    const unsigned short* ab = Aih + ((size_t)(((d * MT + mtb) * KT_IH + kt) * 32 + lane)) * 16;
    const size_t mstep = (size_t)KT_IH * 32 * 16;     // one M-tile of fragments
    const uint4* pa0 = (const uint4*)(ab);
    const uint4* pa1 = (const uint4*)(ab + mstep);
    const uint4* pa2 = (const uint4*)(ab + 2 * mstep);
    const uint4* pa3 = (const uint4*)(ab + 3 * mstep);
    a0.q[0] = pa0[0]; a0.q[1] = pa0[1];
    a1.q[0] = pa1[0]; a1.q[1] = pa1[1];
    a2.q[0] = pa2[0]; a2.q[1] = pa2[1];
    a3.q[0] = pa3[0]; a3.q[1] = pa3[1];
    acc0 = __builtin_amdgcn_wmma_f32_16x16x32_bf16(false, a0.v, false, bf_.v, (short)0, acc0, false, false);
    acc1 = __builtin_amdgcn_wmma_f32_16x16x32_bf16(false, a1.v, false, bf_.v, (short)0, acc1, false, false);
    acc2 = __builtin_amdgcn_wmma_f32_16x16x32_bf16(false, a2.v, false, bf_.v, (short)0, acc2, false, false);
    acc3 = __builtin_amdgcn_wmma_f32_16x16x32_bf16(false, a3.v, false, bf_.v, (short)0, acc3, false, false);
  }

  // D layout: VGPR r -> M = mt*16 + r (+8 for lanes 16-31); N = lane%16.
  float* gIt = gI + (size_t)(t * 2 + d) * G4 * COLS;
  const int mrow = (lane < 16) ? 0 : 8;
#pragma unroll
  for (int r = 0; r < 8; ++r) {
    gIt[(size_t)((mtb + 0) * 16 + r + mrow) * COLS + col] = acc0[r];
    gIt[(size_t)((mtb + 1) * 16 + r + mrow) * COLS + col] = acc1[r];
    gIt[(size_t)((mtb + 2) * 16 + r + mrow) * COLS + col] = acc2[r];
    gIt[(size_t)((mtb + 3) * 16 + r + mrow) * COLS + col] = acc3[r];
  }
}

// ---------------------------------------------------------------------------
// 5) One recurrence step: gh = W_hh * h_{t-1} (WMMA), gates -> LDS,
//    LSTM pointwise, h_t stored bf16.
// ---------------------------------------------------------------------------
__global__ void __launch_bounds__(256) k_scan(const unsigned short* __restrict__ Ahh,
                                              const float* __restrict__ gI,
                                              unsigned short* __restrict__ hbuf,
                                              float* __restrict__ cbuf,
                                              const float* __restrict__ bih_f,
                                              const float* __restrict__ bhh_f,
                                              const float* __restrict__ bih_b,
                                              const float* __restrict__ bhh_b,
                                              int t) {
  __shared__ float gbuf[256][16];                     // gates for this d, 16 cols
  const int nt = blockIdx.x, d = blockIdx.y;
  const int tid = threadIdx.x;
  const int wave = tid >> 5, lane = tid & 31;
  const int mt0 = wave * 2, mt1 = wave * 2 + 1;
  const int col = nt * 16 + (lane & 15);
  const int kbase = (lane < 16) ? 0 : 16;

  v8f acc0 = {}, acc1 = {};
  if (t > 0) {
    const unsigned short* hprev = hbuf + (size_t)((t - 1) * 2 + d) * HID * COLS;
#pragma unroll
    for (int kt = 0; kt < 2; ++kt) {
      Frag bf_;
#pragma unroll
      for (int e = 0; e < 16; ++e) {
        int Kg = kt * 32 + kbase + e;                 // hidden index < 64
        bf_.s[e] = hprev[(size_t)Kg * COLS + col];
      }
      Frag a0, a1;
      const uint4* pa0 = (const uint4*)(Ahh + ((size_t)(((d * MT + mt0) * 2 + kt) * 32 + lane)) * 16);
      const uint4* pa1 = (const uint4*)(Ahh + ((size_t)(((d * MT + mt1) * 2 + kt) * 32 + lane)) * 16);
      a0.q[0] = pa0[0]; a0.q[1] = pa0[1];
      a1.q[0] = pa1[0]; a1.q[1] = pa1[1];
      acc0 = __builtin_amdgcn_wmma_f32_16x16x32_bf16(false, a0.v, false, bf_.v, (short)0, acc0, false, false);
      acc1 = __builtin_amdgcn_wmma_f32_16x16x32_bf16(false, a1.v, false, bf_.v, (short)0, acc1, false, false);
    }
  }
  const int mrow = (lane < 16) ? 0 : 8;
#pragma unroll
  for (int r = 0; r < 8; ++r) {
    gbuf[mt0 * 16 + r + mrow][lane & 15] = acc0[r];
    gbuf[mt1 * 16 + r + mrow][lane & 15] = acc1[r];
  }
  __syncthreads();

  // LSTM pointwise: 64 hidden x 16 cols = 1024 cells, 4 per thread.
  const float* gIt = gI + (size_t)(t * 2 + d) * G4 * COLS;
  const float* bih = d ? bih_b : bih_f;
  const float* bhh = d ? bhh_b : bhh_f;
#pragma unroll
  for (int q = 0; q < 4; ++q) {
    int idx = tid * 4 + q;
    int hh = idx >> 4, cc = idx & 15;
    int pc = nt * 16 + cc;
    int ri = hh, rf = 64 + hh, rg = 128 + hh, ro = 192 + hh;
    float gi = gbuf[ri][cc] + gIt[(size_t)ri * COLS + pc] + bih[ri] + bhh[ri];
    float gf = gbuf[rf][cc] + gIt[(size_t)rf * COLS + pc] + bih[rf] + bhh[rf];
    float gg = gbuf[rg][cc] + gIt[(size_t)rg * COLS + pc] + bih[rg] + bhh[rg];
    float go = gbuf[ro][cc] + gIt[(size_t)ro * COLS + pc] + bih[ro] + bhh[ro];
    float cold = (t == 0) ? 0.0f : cbuf[(size_t)(d * HID + hh) * COLS + pc];
    float cnew = sigm(gf) * cold + sigm(gi) * tanh_f(gg);
    float hnew = sigm(go) * tanh_f(cnew);
    cbuf[(size_t)(d * HID + hh) * COLS + pc] = cnew;
    hbuf[(size_t)((t * 2 + d) * HID + hh) * COLS + pc] = f2bf(hnew);
  }
}

// ---------------------------------------------------------------------------
// 6) Transposed conv (stride 2, kernel 8) + double PReLU + residual.
// ---------------------------------------------------------------------------
__global__ void __launch_bounds__(128) k_proj(const unsigned short* __restrict__ hbuf,
                                              const float* __restrict__ Wp,
                                              const float* __restrict__ bp,
                                              const float* __restrict__ pra,
                                              const float* __restrict__ x,
                                              float* __restrict__ out) {
  __shared__ float sW[128 * 8];                       // W_proj[:, c, :]
  __shared__ float sH[128 * 61];                      // h[t, :, n*61 .. +61)
  const int t = blockIdx.x, n = blockIdx.y, c = blockIdx.z;
  const int tid = threadIdx.x;

#pragma unroll
  for (int r = 0; r < 8; ++r) {
    int i = tid + r * 128;                            // 1024 values
    int m = i >> 3, k = i & 7;
    sW[i] = Wp[((size_t)m * C + c) * WINW + k];
  }
  for (int i = tid; i < 128 * L; i += 128) {
    int m = i / L, l = i - m * L;
    sH[i] = bf2f(hbuf[(size_t)((t * 2 + (m >> 6)) * HID + (m & 63)) * COLS + n * L + l]);
  }
  __syncthreads();

  const int f = tid;
  float acc = bp[c];
  const int k0 = f & 1;
#pragma unroll
  for (int j = 0; j < 4; ++j) {
    int k = k0 + 2 * j;
    int l = (f - k) >> 1;                             // f = 2l + k
    if (l >= 0 && l < L) {
      const float* hl = &sH[l];
#pragma unroll 4
      for (int m = 0; m < 128; ++m)
        acc += hl[m * L] * sW[m * 8 + k];
    }
  }
  float a = pra[0];
  float p = (acc >= 0.0f) ? acc : a * acc;            // prelu #1
  p = (p >= 0.0f) ? p : a * p;                        // prelu #2
  int b = n >> 3, w = n & 7, tg = t * 8 + w;
  size_t o = (((size_t)b * C + c) * T + tg) * Ff + f;
  out[o] = p + x[o];                                  // residual
}

// ---------------------------------------------------------------------------
// Host-side orchestration
// ---------------------------------------------------------------------------
extern "C" void kernel_launch(void* const* d_in, const int* in_sizes, int n_in,
                              void* d_out, int out_size, void* d_ws, size_t ws_size,
                              hipStream_t stream) {
  (void)in_sizes; (void)n_in; (void)out_size; (void)ws_size;
  const float* x      = (const float*)d_in[0];
  const float* gln_g  = (const float*)d_in[1];
  const float* gln_b  = (const float*)d_in[2];
  const float* pra    = (const float*)d_in[3];
  const float* W_ih_f = (const float*)d_in[4];
  const float* b_ih_f = (const float*)d_in[5];
  const float* W_hh_f = (const float*)d_in[6];
  const float* b_hh_f = (const float*)d_in[7];
  const float* W_ih_b = (const float*)d_in[8];
  const float* b_ih_b = (const float*)d_in[9];
  const float* W_hh_b = (const float*)d_in[10];
  const float* b_hh_b = (const float*)d_in[11];
  const float* W_proj = (const float*)d_in[12];
  const float* b_proj = (const float*)d_in[13];
  float* out = (float*)d_out;

  char* ws = (char*)d_ws;
  const size_t OFF_STATS = 0;                         // 8 f32
  const size_t OFF_XS    = 256;                       // 16,777,216 bf16 = 32 MB
  const size_t OFF_AIH   = OFF_XS  + (size_t)16777216 * 2;
  const size_t OFF_AHH   = OFF_AIH + (size_t)1310720 * 2;
  const size_t OFF_GI    = OFF_AHH + (size_t)32768 * 2;
  const size_t OFF_H     = OFF_GI  + (size_t)NW * 2 * G4 * COLS * 4;   // 128 MB
  const size_t OFF_C     = OFF_H   + (size_t)NW * 2 * HID * COLS * 2;  // 16 MB

  float*          stats = (float*)(ws + OFF_STATS);
  unsigned short* xs    = (unsigned short*)(ws + OFF_XS);
  unsigned short* Aih   = (unsigned short*)(ws + OFF_AIH);
  unsigned short* Ahh   = (unsigned short*)(ws + OFF_AHH);
  float*          gI    = (float*)(ws + OFF_GI);
  unsigned short* hbuf  = (unsigned short*)(ws + OFF_H);
  float*          cbuf  = (float*)(ws + OFF_C);

  hipMemsetAsync(stats, 0, 8 * sizeof(float), stream);
  k_stats<<<dim3(1024, Bb), 256, 0, stream>>>(x, stats);
  k_xs<<<dim3(NW * Nn), 256, 0, stream>>>(x, gln_g, gln_b, stats, xs);
  k_swz_ih<<<dim3(5120), 256, 0, stream>>>(W_ih_f, W_ih_b, Aih);
  k_swz_hh<<<dim3(128), 256, 0, stream>>>(W_hh_f, W_hh_b, Ahh);
  k_gates<<<dim3(NT32, NW, 2), 256, 0, stream>>>(xs, Aih, gI);
  for (int t = 0; t < NW; ++t)
    k_scan<<<dim3(COLS / 16, 2), 256, 0, stream>>>(Ahh, gI, hbuf, cbuf,
                                                   b_ih_f, b_hh_f, b_ih_b, b_hh_b, t);
  k_proj<<<dim3(NW, Nn, C), 128, 0, stream>>>(hbuf, W_proj, b_proj, pra, x, out);
}
